// GraphConv_10926396801181
// MI455X (gfx1250) — matrix operand order repaired and checked by reading, not compiled, not run on hardware
//
#include <hip/hip_runtime.h>

#define B_ 16
#define N_ 2048
#define D_ 64

typedef __attribute__((ext_vector_type(16))) __bf16 v16bf;
typedef __attribute__((ext_vector_type(8)))  float  v8f;
typedef __attribute__((ext_vector_type(4)))  float  v4f;

static __device__ __forceinline__ v8f wmma_bf16(v16bf a, v16bf b, v8f c) {
    return __builtin_amdgcn_wmma_f32_16x16x32_bf16(
        /*neg_a=*/false, a, /*neg_b=*/false, b,
        /*c_mod=*/(short)0, c, /*reuse_a=*/false, /*reuse_b=*/false);
}

// ---------------------------------------------------------------------------
// Phase 1: degree[b,n] = sum_m adj[b,n,m];  dinv = rsqrt(degree) (0 if deg==0)
// One wave32 per row; non-temporal float4 streaming; shfl_xor reduction.
// ---------------------------------------------------------------------------
__global__ void __launch_bounds__(256)
degree_kernel(const float* __restrict__ adj, float* __restrict__ dinv) {
    const int wave = threadIdx.x >> 5;
    const int lane = threadIdx.x & 31;
    const int row  = blockIdx.x * 8 + wave;           // [0, B_*N_)
    const float* r = adj + (size_t)row * N_;
    float s = 0.f;
#pragma unroll
    for (int i = 0; i < N_ / 128; ++i) {
        v4f v = __builtin_nontemporal_load(
            reinterpret_cast<const v4f*>(r + i * 128 + lane * 4));
        s += v[0] + v[1] + v[2] + v[3];
    }
#pragma unroll
    for (int off = 16; off > 0; off >>= 1)
        s += __shfl_xor(s, off, 32);
    if (lane == 0)
        dinv[row] = (s > 0.f) ? rsqrtf(s) : 0.f;
}

// ---------------------------------------------------------------------------
// Phase 2: sT[b][o][m] = bf16( dinv[b,m] * sum_i x[b,m,i] * W[i,o] )
// Block of 256 handles 4 rows; x rows staged in LDS; W broadcast from L2.
// Stored TRANSPOSED so phase-3 B-operand loads are contiguous along m.
// ---------------------------------------------------------------------------
__global__ void __launch_bounds__(256)
support_kernel(const float* __restrict__ x, const float* __restrict__ W,
               const float* __restrict__ dinv, __bf16* __restrict__ sT) {
    __shared__ float xs[4 * D_];
    const int b   = blockIdx.x >> 9;                  // 512 blocks per batch
    const int m0  = (blockIdx.x & 511) * 4;
    const int tid = threadIdx.x;
    xs[tid] = x[((size_t)b * N_ + m0) * D_ + tid];    // 4 rows x 64 = 256 floats
    __syncthreads();
    const int rl = tid >> 6;                          // row-local 0..3
    const int o  = tid & 63;                          // output feature
    const float* xr = xs + rl * D_;
    float acc = 0.f;
#pragma unroll
    for (int i = 0; i < D_; ++i)
        acc = fmaf(xr[i], W[i * D_ + o], acc);
    acc *= dinv[b * N_ + m0 + rl];
    sT[((size_t)b * D_ + o) * N_ + m0 + rl] = (__bf16)acc;
}

// ---------------------------------------------------------------------------
// Phase 3: out[b,n,o] = relu( dinv[b,n] * sum_m adj[b,n,m]*sT[b,o,m] + bias[o] )
// 8 waves/block; each wave owns a 16-row x 64-col output tile.
// Software-pipelined (ping-pong) inner loop: next K-step's adj (NT, HBM) and
// sT (RT, L2) loads are in flight while current WMMAs execute.
// ---------------------------------------------------------------------------
__global__ void __launch_bounds__(256)
gcn_gemm_kernel(const float* __restrict__ adj, const __bf16* __restrict__ sT,
                const float* __restrict__ dinv, const float* __restrict__ bias,
                float* __restrict__ out) {
    const int b    = blockIdx.y;
    const int wave = threadIdx.x >> 5;
    const int lane = threadIdx.x & 31;
    const int row0 = blockIdx.x * 128 + wave * 16;
    const int half = lane >> 4;
    const int mrow = lane & 15;
    const int h8   = half * 8;                        // C/D row offset
    const int h16  = half * 16;                       // B-operand K offset

    const float*  arow  = adj + ((size_t)b * N_ + row0 + mrow) * N_;
    const __bf16* sbase = sT  + (size_t)b * D_ * N_;

    v8f acc[4] = {};                                  // 4 col-tiles x 16x16 f32

    // A tile: 16x32 fp32, contiguous K layout (matches B):
    //   lane<16  holds M=lane,    K in {0..15}
    //   lane>=16 holds M=lane-16, K in {16..31}
    auto loadA = [&](v4f (&f)[4], int m0) {
        f[0] = __builtin_nontemporal_load((const v4f*)(arow + m0 + h16));
        f[1] = __builtin_nontemporal_load((const v4f*)(arow + m0 + h16 + 4));
        f[2] = __builtin_nontemporal_load((const v4f*)(arow + m0 + h16 + 8));
        f[3] = __builtin_nontemporal_load((const v4f*)(arow + m0 + h16 + 12));
    };
    // B tiles: 32x16 bf16; lanes 0-15 hold K=0..15, lanes 16-31 K=16..31.
    auto loadB = [&](v16bf (&bt)[4], int m0) {
#pragma unroll
        for (int t = 0; t < 4; ++t)
            bt[t] = *reinterpret_cast<const v16bf*>(
                sbase + (size_t)(t * 16 + mrow) * N_ + m0 + h16);
    };
    auto step = [&](const v4f (&f)[4], const v16bf (&bt)[4]) {
        v16bf a;
#pragma unroll
        for (int q = 0; q < 4; ++q)
#pragma unroll
            for (int j = 0; j < 4; ++j)
                a[q * 4 + j] = (__bf16)f[q][j];
#pragma unroll
        for (int t = 0; t < 4; ++t)
            acc[t] = wmma_bf16(a, bt[t], acc[t]);
    };

    v4f   fa0[4], fa1[4];
    v16bf bt0[4], bt1[4];

    loadA(fa0, 0);
    loadB(bt0, 0);
    // Steady state: branch-free ping-pong over 64-wide super-steps.
    for (int m0 = 0; m0 < N_ - 64; m0 += 64) {
        loadA(fa1, m0 + 32);
        loadB(bt1, m0 + 32);
        step(fa0, bt0);
        loadA(fa0, m0 + 64);
        loadB(bt0, m0 + 64);
        step(fa1, bt1);
    }
    // Tail: last 64 columns (buffers fa0/bt0 hold N_-64; fetch N_-32).
    loadA(fa1, N_ - 32);
    loadB(bt1, N_ - 32);
    step(fa0, bt0);
    step(fa1, bt1);

    // --- Epilogue: row-scale by dinv[n], +bias, ReLU, store ---
    // C/D layout: VGPR r holds M = half*8 + r, column N = lane&15.
    const float* dr = dinv + b * N_ + row0 + h8;
    v4f d0 = *reinterpret_cast<const v4f*>(dr);
    v4f d1 = *reinterpret_cast<const v4f*>(dr + 4);
    const float dv[8] = {d0[0], d0[1], d0[2], d0[3], d1[0], d1[1], d1[2], d1[3]};

#pragma unroll
    for (int t = 0; t < 4; ++t) {
        const int col = t * 16 + mrow;
        const float bb = bias[col];
#pragma unroll
        for (int r = 0; r < 8; ++r) {
            float v = acc[t][r] * dv[r] + bb;
            v = (v > 0.f) ? v : 0.f;
            out[((size_t)b * N_ + row0 + h8 + r) * D_ + col] = v;
        }
    }
}

// ---------------------------------------------------------------------------
extern "C" void kernel_launch(void* const* d_in, const int* in_sizes, int n_in,
                              void* d_out, int out_size, void* d_ws, size_t ws_size,
                              hipStream_t stream) {
    const float* x    = (const float*)d_in[0];   // (16, 2048, 64)
    const float* adj  = (const float*)d_in[1];   // (16, 2048, 2048)
    const float* W    = (const float*)d_in[2];   // (64, 64)
    const float* bias = (const float*)d_in[3];   // (64,)
    float* out = (float*)d_out;                  // (16, 2048, 64)

    float*  dinv = (float*)d_ws;                                             // 128 KB
    __bf16* sT   = (__bf16*)((char*)d_ws + (size_t)B_ * N_ * sizeof(float)); // 4 MB

    degree_kernel <<<B_ * N_ / 8, 256, 0, stream>>>(adj, dinv);
    support_kernel<<<B_ * N_ / 4, 256, 0, stream>>>(x, W, dinv, sT);
    gcn_gemm_kernel<<<dim3(N_ / 128, B_), 256, 0, stream>>>(adj, sT, dinv, bias, out);
}
